// CrystalGraphConvNet_79817672229309
// MI455X (gfx1250) — compile-verified, matrix operand order checked by reading, not run on hardware
//
#include <hip/hip_runtime.h>
#include <hip/hip_bf16.h>

// ---------------------------------------------------------------------------
// CrystalGraphConvNet VAE forward for gfx1250 (MI455X).
// Heavy einsums -> v_wmma_f32_16x16x32_bf16 (fp32 accumulate).
// ---------------------------------------------------------------------------

typedef __bf16 bf16_t;
typedef __attribute__((ext_vector_type(8)))  __bf16 v8bf;
typedef __attribute__((ext_vector_type(16))) __bf16 v16bf;
typedef __attribute__((ext_vector_type(8)))  float  v8f;

#define DN1   12000
#define DM1   12
#define DL    92
#define DONB  41
#define DA    128
#define DB    64
#define DN2   300
#define DAP   40
#define DNM   (DN1*DM1)      /* 144000 */
#define DKT   (2*DA+DB)      /* 320    */
#define DOUTW (2*DL+DONB)    /* 225    */

__device__ __forceinline__ float sp_(float x){
  return x > 0.f ? x + log1pf(__expf(-x)) : log1pf(__expf(x));
}
__device__ __forceinline__ float sg_(float x){ return 1.f/(1.f+__expf(-x)); }

// ---------------------------------------------------------------------------
// WMMA GEMM:  C[M x Ncols] = X[M x Kpad](bf16,row-major) * W[Npad x Kpad]^T + bias
// ISA 7.12.2 16-bit fragment layout: per lane two contiguous 8-element runs.
// ---------------------------------------------------------------------------
__device__ __forceinline__ v16bf ldfrag(const bf16_t* p){
  v8bf lo = *(const v8bf*)(p);
  v8bf hi = *(const v8bf*)(p + 16);
  return __builtin_shufflevector(lo, hi, 0,1,2,3,4,5,6,7,8,9,10,11,12,13,14,15);
}

__global__ __launch_bounds__(128)
void k_gemm(const bf16_t* __restrict__ X, const bf16_t* __restrict__ W,
            float* __restrict__ C, const float* __restrict__ bias,
            int M, int Ncols, int Kpad)
{
  const int wave = threadIdx.x >> 5;
  const int lane = threadIdx.x & 31;
  const int m0   = (blockIdx.x * 4 + wave) * 16;
  if (m0 >= M) return;                       // whole-wave exit: EXEC stays all-1s
  const int n0   = blockIdx.y * 64;
  const int half = lane >> 4;
  const int rm   = lane & 15;

  const bf16_t* ap = X + (size_t)(m0 + rm) * Kpad + half * 8;
  const bf16_t* bp = W + (size_t)(n0 + rm) * Kpad + half * 8;
  const size_t wstride = (size_t)16 * Kpad;

  v8f a0 = {0,0,0,0,0,0,0,0}, a1 = a0, a2 = a0, a3 = a0;
  for (int k = 0; k < Kpad; k += 32) {
    v16bf af = ldfrag(ap + k);
    v16bf b0 = ldfrag(bp + k);
    v16bf b1 = ldfrag(bp + wstride   + k);
    v16bf b2 = ldfrag(bp + 2*wstride + k);
    v16bf b3 = ldfrag(bp + 3*wstride + k);
    a0 = __builtin_amdgcn_wmma_f32_16x16x32_bf16(false, af, false, b0, (short)0, a0, false, false);
    a1 = __builtin_amdgcn_wmma_f32_16x16x32_bf16(false, af, false, b1, (short)0, a1, false, false);
    a2 = __builtin_amdgcn_wmma_f32_16x16x32_bf16(false, af, false, b2, (short)0, a2, false, false);
    a3 = __builtin_amdgcn_wmma_f32_16x16x32_bf16(false, af, false, b3, (short)0, a3, false, false);
  }
  v8f acc[4] = {a0, a1, a2, a3};
#pragma unroll
  for (int t = 0; t < 4; ++t) {
    int col = n0 + t*16 + rm;
    if (col >= Ncols) continue;
    float bv = bias ? bias[col] : 0.f;
    float* cp = C + (size_t)(m0 + half*8) * Ncols + col;
#pragma unroll
    for (int v = 0; v < 8; ++v)
      cp[(size_t)v * Ncols] = acc[t][v] + bv;
  }
}

// ---------------------------------------------------------------------------
// Elementwise / helper kernels
// ---------------------------------------------------------------------------
__global__ void k_total_input(const float* __restrict__ atom, const float* __restrict__ nbr,
                              const int* __restrict__ idx, float* __restrict__ out)
{
  long long i = (long long)blockIdx.x * blockDim.x + threadIdx.x;
  if (i >= (long long)DNM * DOUTW) return;
  int r = (int)(i / DOUTW), t = (int)(i % DOUTW);
  int n = r / DM1;
  float v;
  if (t < DL)            v = atom[(size_t)n * DL + t];
  else if (t < 2*DL)     v = atom[(size_t)idx[r] * DL + (t - DL)];
  else                   v = nbr[(size_t)r * DONB + (t - 2*DL)];
  out[i] = v;
}

__global__ void k_conv_x(const float* __restrict__ src, bf16_t* __restrict__ dst,
                         int rows, int K, int Kpad)
{
  long long i = (long long)blockIdx.x * blockDim.x + threadIdx.x;
  if (i >= (long long)rows * Kpad) return;
  int r = (int)(i / Kpad), k = (int)(i % Kpad);
  dst[i] = (k < K) ? (bf16_t)src[(size_t)r * K + k] : (bf16_t)0.f;
}

__global__ void k_conv_w(const float* __restrict__ src, bf16_t* __restrict__ dst,
                         int N, int K, int Npad, int Kpad)
{
  long long i = (long long)blockIdx.x * blockDim.x + threadIdx.x;
  if (i >= (long long)Npad * Kpad) return;
  int n = (int)(i / Kpad), k = (int)(i % Kpad);
  dst[i] = (n < N && k < K) ? (bf16_t)src[(size_t)n * K + k] : (bf16_t)0.f;
}

__global__ void k_build_tot_conv(const float* __restrict__ a, const float* __restrict__ nb,
                                 const int* __restrict__ idx, bf16_t* __restrict__ tot)
{
  long long i = (long long)blockIdx.x * blockDim.x + threadIdx.x;
  if (i >= (long long)DNM * DKT) return;
  int r = (int)(i / DKT), k = (int)(i % DKT);
  int n = r / DM1;
  float v;
  if (k < DA)          v = a[(size_t)n * DA + k];
  else if (k < 2*DA)   v = a[(size_t)idx[r] * DA + (k - DA)];
  else                 v = nb[(size_t)r * DB + (k - 2*DA)];
  tot[i] = (bf16_t)v;
}

__global__ void k_build_tot_deconv(const float* __restrict__ a, const float* __restrict__ zan,
                                   const float* __restrict__ dnb, bf16_t* __restrict__ tot)
{
  long long i = (long long)blockIdx.x * blockDim.x + threadIdx.x;
  if (i >= (long long)DNM * DKT) return;
  int r = (int)(i / DKT), k = (int)(i % DKT);
  int n = r / DM1;
  float v;
  if (k < DA)          v = a[(size_t)n * DA + k];
  else if (k < 2*DA)   v = zan[(size_t)r * DA + (k - DA)];
  else                 v = dnb[(size_t)r * DB + (k - 2*DA)];
  tot[i] = (bf16_t)v;
}

// one block per column: deterministic mean/var reduction
__global__ void k_bn_stats(const float* __restrict__ x, int rows, int cols,
                           float* __restrict__ sums)
{
  int c = blockIdx.x, t = threadIdx.x;
  float s = 0.f, q = 0.f;
  for (int r = t; r < rows; r += blockDim.x) {
    float v = x[(size_t)r * cols + c];
    s += v; q += v * v;
  }
  __shared__ float sh[256], shq[256];
  sh[t] = s; shq[t] = q; __syncthreads();
  for (int o = 128; o > 0; o >>= 1) {
    if (t < o) { sh[t] += sh[t+o]; shq[t] += shq[t+o]; }
    __syncthreads();
  }
  if (t == 0) { sums[c] = sh[0]; sums[256 + c] = shq[0]; }
}

__global__ void k_bn_finalize(const float* __restrict__ sums, float rows,
                              const float* __restrict__ g, const float* __restrict__ b,
                              int cols, float* __restrict__ scale, float* __restrict__ shift)
{
  int c = threadIdx.x;
  if (c >= cols) return;
  float mean = sums[c] / rows;
  float var  = fmaxf(sums[256 + c] / rows - mean * mean, 0.f);
  float s    = g[c] * rsqrtf(var + 1e-5f);
  scale[c] = s;
  shift[c] = b[c] - mean * s;
}

// a-branch: gate + sum over M1 neighbors  -> asum[N1 x A]
__global__ void k_gate_asum(const float* __restrict__ ga, const float* __restrict__ sc,
                            const float* __restrict__ sh, float* __restrict__ asum)
{
  long long i = (long long)blockIdx.x * blockDim.x + threadIdx.x;
  if (i >= (long long)DN1 * DA) return;
  int n = (int)(i / DA), c = (int)(i % DA);
  float acc = 0.f;
  for (int m = 0; m < DM1; ++m) {
    size_t row = (size_t)(n * DM1 + m) * (2*DA);
    float x1 = ga[row + c]       * sc[c]       + sh[c];
    float x2 = ga[row + DA + c]  * sc[DA + c]  + sh[DA + c];
    acc += sg_(x1) * sp_(x2);
  }
  asum[i] = acc;
}

// generic gate: h[r][c] = sg(norm(g[r][c])) * sp(norm(g[r][C+c]))
__global__ void k_gate(const float* __restrict__ g, const float* __restrict__ sc,
                       const float* __restrict__ sh, float* __restrict__ h,
                       int rows, int C)
{
  long long i = (long long)blockIdx.x * blockDim.x + threadIdx.x;
  if (i >= (long long)rows * C) return;
  int r = (int)(i / C), c = (int)(i % C);
  size_t row = (size_t)r * (2 * C);
  float x1 = g[row + c]     * sc[c]     + sh[c];
  float x2 = g[row + C + c] * sc[C + c] + sh[C + c];
  h[i] = sg_(x1) * sp_(x2);
}

__global__ void k_residual_sp(float* __restrict__ base, const float* __restrict__ h,
                              const float* __restrict__ sc, const float* __restrict__ sh,
                              long long n, int C)
{
  long long i = (long long)blockIdx.x * blockDim.x + threadIdx.x;
  if (i >= n) return;
  int c = (int)(i % C);
  base[i] = sp_(base[i] + h[i] * sc[c] + sh[c]);
}

__global__ void k_pool_a(const float* __restrict__ a, float* __restrict__ pooled)
{
  long long i = (long long)blockIdx.x * blockDim.x + threadIdx.x;
  if (i >= (long long)DN2 * DA) return;
  int n2 = (int)(i / DA), c = (int)(i % DA);
  float s = 0.f;
  for (int k = 0; k < DAP; ++k) s += a[(size_t)(n2 * DAP + k) * DA + c];
  pooled[(size_t)n2 * 896 + c] = sp_(s / (float)DAP);
}

__global__ void k_pool_nb(const float* __restrict__ nb, float* __restrict__ pooled)
{
  long long i = (long long)blockIdx.x * blockDim.x + threadIdx.x;
  if (i >= (long long)DN2 * (DM1*DB)) return;
  int n2 = (int)(i / (DM1*DB)), q = (int)(i % (DM1*DB));
  int m = q / DB, b = q % DB;
  float s = 0.f;
  for (int k = 0; k < DAP; ++k)
    s += nb[(size_t)((n2 * DAP + k) * DM1 + m) * DB + b];
  pooled[(size_t)n2 * 896 + DA + q] = sp_(s / (float)DAP);
}

__global__ void k_nbv(const float* __restrict__ pooled, const float* __restrict__ pW,
                      const float* __restrict__ pB, float* __restrict__ nbv)
{
  long long i = (long long)blockIdx.x * blockDim.x + threadIdx.x;
  if (i >= (long long)DN2 * DB) return;
  int n2 = (int)(i / DB), b = (int)(i % DB);
  float s = pB[0];
  for (int j = 0; j < DM1; ++j)
    s += pooled[(size_t)n2 * 896 + DA + b * DM1 + j] * pW[j];   // nbp viewed as (B,M1)
  nbv[i] = s;
}

__global__ void k_build_crys(const float* __restrict__ pooled, const float* __restrict__ nbv,
                             const float* __restrict__ sc, const float* __restrict__ sh,
                             float* __restrict__ crys)
{
  long long i = (long long)blockIdx.x * blockDim.x + threadIdx.x;
  if (i >= (long long)DN2 * (DA + DB)) return;
  int n2 = (int)(i / (DA + DB)), k = (int)(i % (DA + DB));
  crys[i] = (k < DA) ? pooled[(size_t)n2 * 896 + k]
                     : nbv[(size_t)n2 * DB + (k - DA)] * sc[k - DA] + sh[k - DA];
}

__global__ void k_vae(const float* __restrict__ crys,
                      const float* __restrict__ muW, const float* __restrict__ muB,
                      const float* __restrict__ lvW, const float* __restrict__ lvB,
                      const float* __restrict__ eps,
                      float* __restrict__ omu, float* __restrict__ olv,
                      float* __restrict__ oz, float* __restrict__ zbuf)
{
  long long i = (long long)blockIdx.x * blockDim.x + threadIdx.x;
  if (i >= (long long)DN2 * 2) return;
  int n2 = (int)(i / 2), d = (int)(i % 2);
  float mu = muB[d], lv = lvB[d];
  for (int k = 0; k < (DA + DB); ++k) {
    float c = crys[(size_t)n2 * (DA + DB) + k];
    mu += c * muW[(size_t)d * (DA + DB) + k];
    lv += c * lvW[(size_t)d * (DA + DB) + k];
  }
  float z = mu + eps[i] * __expf(0.5f * lv);
  omu[i] = mu; olv[i] = lv; oz[i] = z; zbuf[i] = z;
}

__global__ void k_zd(const float* __restrict__ zbuf, const float* __restrict__ dW,
                     const float* __restrict__ dB, float* __restrict__ zdcat,
                     float* __restrict__ zdn)
{
  long long i = (long long)blockIdx.x * blockDim.x + threadIdx.x;
  if (i >= (long long)DN2 * (DA + DB)) return;
  int n2 = (int)(i / (DA + DB)), k = (int)(i % (DA + DB));
  float s = dB[k];
  s += zbuf[(size_t)n2 * 2 + 0] * dW[(size_t)k * 2 + 0];
  s += zbuf[(size_t)n2 * 2 + 1] * dW[(size_t)k * 2 + 1];
  if (k < DA) zdcat[(size_t)n2 * 896 + k] = s;
  else        zdn[(size_t)n2 * DB + (k - DA)] = s;
}

__global__ void k_zdn2(const float* __restrict__ zdn, const float* __restrict__ pW2,
                       const float* __restrict__ pB2, float* __restrict__ zdn2)
{
  long long i = (long long)blockIdx.x * blockDim.x + threadIdx.x;
  if (i >= (long long)DN2 * (DB * DM1)) return;
  int n2 = (int)(i / (DB * DM1)), q = (int)(i % (DB * DM1));
  int b = q / DM1, j = q % DM1;
  zdn2[i] = zdn[(size_t)n2 * DB + b] * pW2[j] + pB2[j];
}

__global__ void k_apply_zdn2(const float* __restrict__ zdn2, const float* __restrict__ sc,
                             const float* __restrict__ sh, float* __restrict__ zdcat)
{
  long long i = (long long)blockIdx.x * blockDim.x + threadIdx.x;
  if (i >= (long long)DN2 * 768) return;
  int c = (int)(i % DB);                // (3600 x 64) view column
  int n2 = (int)(i / 768), q = (int)(i % 768);
  zdcat[(size_t)n2 * 896 + DA + q] = zdn2[i] * sc[c] + sh[c];
}

__global__ void k_dec(const float* __restrict__ a, const float* __restrict__ nb,
                      const float* __restrict__ zdcat, const float* __restrict__ pooled,
                      float* __restrict__ zda2, float* __restrict__ dnb)
{
  long long i = (long long)blockIdx.x * blockDim.x + threadIdx.x;
  if (i >= (long long)DN1 * 896) return;
  int n1 = (int)(i / 896), k = (int)(i % 896);
  int n2 = n1 / DAP;
  float npv = (k < DA) ? a[(size_t)n1 * DA + k] : nb[(size_t)n1 * 768 + (k - DA)];
  float v = sp_(npv * zdcat[(size_t)n2 * 896 + k] / pooled[(size_t)n2 * 896 + k]);
  if (k < DA) zda2[(size_t)n1 * DA + k] = v;
  else        dnb[(size_t)n1 * 768 + (k - DA)] = v;
}

__global__ void k_bcast_zan(const float* __restrict__ tmp, float* __restrict__ zan)
{
  long long i = (long long)blockIdx.x * blockDim.x + threadIdx.x;
  if (i >= (long long)DNM * DA) return;
  int r = (int)(i / DA), c = (int)(i % DA);
  zan[i] = tmp[(size_t)(r / DM1) * DA + c];
}

__global__ void k_emit(const float* __restrict__ zr, const float* __restrict__ zanl,
                       const float* __restrict__ dnbl, float* __restrict__ out)
{
  long long i = (long long)blockIdx.x * blockDim.x + threadIdx.x;
  if (i >= (long long)DNM * DOUTW) return;
  int r = (int)(i / DOUTW), t = (int)(i % DOUTW);
  int n = r / DM1;
  float v;
  if (t < DL)        v = zr[(size_t)n * DL + t];
  else if (t < 2*DL) v = zanl[(size_t)r * DL + (t - DL)];
  else               v = dnbl[(size_t)r * DONB + (t - 2*DL)];
  out[i] = sg_(v);
}

// ---------------------------------------------------------------------------
// Host side
// ---------------------------------------------------------------------------
static inline dim3 gs(long long n){ return dim3((unsigned)((n + 255) / 256)); }

static void gemm(const bf16_t* X, const bf16_t* W, float* C, const float* bias,
                 int M, int Ncols, int Npad, int Kpad, hipStream_t s)
{
  dim3 grid((unsigned)((M / 16 + 3) / 4), (unsigned)(Npad / 64));
  k_gemm<<<grid, 128, 0, s>>>(X, W, C, bias, M, Ncols, Kpad);
}

enum {
  I_ATOM=0, I_NBR=1, I_EPS=2,
  I_EMB_ATOM_W=3, I_EMB_ATOM_B=4, I_EMB_ATOM2_W=5, I_EMB_ATOM2_B=6,
  I_EMB_BOND_W=7, I_EMB_BOND_B=8, I_EMB_BOND2_W=9, I_EMB_BOND2_B=10,
  I_PEMB_W=11, I_PEMB_B=12, I_PEMB2_W=13, I_PEMB2_B=14,
  I_BN_G=15, I_BN_B=16, I_MU_W=17, I_MU_B=18, I_LV_W=19, I_LV_B=20,
  I_DEC_W=21, I_DEC_B=22, I_ADJ_W=23, I_ADJ_B=24, I_EMB_NBR_W=25, I_EMB_NBR_B=26,
  I_CONV0=27,    // per conv (+10): Wa,Wb,bn1a_g,bn1a_b,bn1b_g,bn1b_b,bn2a_g,bn2a_b,bn2b_g,bn2b_b
  I_DECONV0=57   // per deconv (+11): same 10 + Wc
};

extern "C" void kernel_launch(void* const* d_in, const int* in_sizes, int n_in,
                              void* d_out, int out_size, void* d_ws, size_t ws_size,
                              hipStream_t stream)
{
  (void)in_sizes; (void)out_size; (void)ws_size;
  auto F = [&](int i)->const float* { return (const float*)d_in[i]; };
  const int* idx = (const int*)d_in[n_in - 1];   // nbr_fea_idx is the last input

  // ---- workspace bump allocator --------------------------------------------
  char* base = (char*)d_ws;
  size_t off = 0;
  auto alloc = [&](size_t bytes)->void* {
    void* p = base + off;
    off = (off + bytes + 255) & ~(size_t)255;
    return p;
  };
  bf16_t* tot    = (bf16_t*)alloc(sizeof(bf16_t)*(size_t)DNM*DKT);
  float*  big256 = (float*) alloc(sizeof(float)*(size_t)DNM*256);
  float*  h128   = (float*) alloc(sizeof(float)*(size_t)DNM*128);
  float*  zan    = (float*) alloc(sizeof(float)*(size_t)DNM*128);
  float*  nb     = (float*) alloc(sizeof(float)*(size_t)DNM*64);
  float*  dnb    = (float*) alloc(sizeof(float)*(size_t)DNM*64);
  float*  a      = (float*) alloc(sizeof(float)*(size_t)DN1*128);
  float*  zda2   = (float*) alloc(sizeof(float)*(size_t)DN1*128);
  float*  asum   = (float*) alloc(sizeof(float)*(size_t)DN1*128);
  float*  pooled = (float*) alloc(sizeof(float)*DN2*896);
  float*  nbv    = (float*) alloc(sizeof(float)*DN2*64);
  float*  crys   = (float*) alloc(sizeof(float)*DN2*192);
  float*  zbuf   = (float*) alloc(sizeof(float)*DN2*2);
  float*  zdcat  = (float*) alloc(sizeof(float)*DN2*896);
  float*  zdn    = (float*) alloc(sizeof(float)*DN2*64);
  float*  zdn2   = (float*) alloc(sizeof(float)*DN2*768);
  float*  stats  = (float*) alloc(sizeof(float)*512);
  float*  sc1    = (float*) alloc(sizeof(float)*256);
  float*  sh1    = (float*) alloc(sizeof(float)*256);
  float*  sc2    = (float*) alloc(sizeof(float)*256);
  float*  sh2    = (float*) alloc(sizeof(float)*256);
  bf16_t* wEmbAtom  = (bf16_t*)alloc(sizeof(bf16_t)*128*96);
  bf16_t* wEmbBond  = (bf16_t*)alloc(sizeof(bf16_t)*64*64);
  bf16_t* wEmbAtom2 = (bf16_t*)alloc(sizeof(bf16_t)*128*128);
  bf16_t* wEmbBond2 = (bf16_t*)alloc(sizeof(bf16_t)*64*64);
  bf16_t* wEmbNbr   = (bf16_t*)alloc(sizeof(bf16_t)*128*128);
  bf16_t* wAdj      = (bf16_t*)alloc(sizeof(bf16_t)*128*128);
  bf16_t *wCa[3], *wCb[3], *wDa[3], *wDb[3], *wDc[3];
  for (int l = 0; l < 3; ++l) {
    wCa[l] = (bf16_t*)alloc(sizeof(bf16_t)*256*DKT);
    wCb[l] = (bf16_t*)alloc(sizeof(bf16_t)*128*DKT);
  }
  for (int l = 0; l < 3; ++l) {
    wDa[l] = (bf16_t*)alloc(sizeof(bf16_t)*256*DKT);
    wDb[l] = (bf16_t*)alloc(sizeof(bf16_t)*128*DKT);
    wDc[l] = (bf16_t*)alloc(sizeof(bf16_t)*256*DKT);
  }
  // bf16 staging aliased into tot (used only while tot is idle)
  bf16_t* xaBf   = tot;                                        // 12000 x 96
  bf16_t* xnbBf  = tot + (size_t)DN1*96;                       // 144000 x 64
  bf16_t* zanBf  = tot;                                        // 144000 x 128
  bf16_t* dnbBf  = tot + (size_t)DNM*128;                      // 144000 x 64
  bf16_t* zda2Bf = tot + (size_t)DNM*128 + (size_t)DNM*64;     // 12000 x 128

  // ---- output layout -------------------------------------------------------
  float* o_out = (float*)d_out;
  float* o_mu  = o_out + (size_t)DNM*DOUTW;
  float* o_lv  = o_mu  + DN2*2;
  float* o_z   = o_lv  + DN2*2;
  float* o_tif = o_z   + DN2*2;

  auto bn_make = [&](const float* x, long long rows, int cols,
                     const float* g, const float* b, float* sc, float* sh){
    k_bn_stats<<<dim3((unsigned)cols), 256, 0, stream>>>(x, (int)rows, cols, stats);
    k_bn_finalize<<<1, 256, 0, stream>>>(stats, (float)rows, g, b, cols, sc, sh);
  };

  // ---- weight conversion (fp32 -> bf16, padded) ----------------------------
  k_conv_w<<<gs(128*96),256,0,stream>>>(F(I_EMB_ATOM_W),  wEmbAtom,  128,  92, 128,  96);
  k_conv_w<<<gs(64*64),256,0,stream>>>( F(I_EMB_BOND_W),  wEmbBond,   64,  41,  64,  64);
  k_conv_w<<<gs(128*128),256,0,stream>>>(F(I_EMB_ATOM2_W),wEmbAtom2,  92, 128, 128, 128);
  k_conv_w<<<gs(64*64),256,0,stream>>>( F(I_EMB_BOND2_W), wEmbBond2,  41,  64,  64,  64);
  k_conv_w<<<gs(128*128),256,0,stream>>>(F(I_EMB_NBR_W),  wEmbNbr,    92, 128, 128, 128);
  k_conv_w<<<gs(128*128),256,0,stream>>>(F(I_ADJ_W),      wAdj,      128, 128, 128, 128);
  for (int l = 0; l < 3; ++l) {
    k_conv_w<<<gs(256*DKT),256,0,stream>>>(F(I_CONV0+10*l+0), wCa[l], 256, DKT, 256, DKT);
    k_conv_w<<<gs(128*DKT),256,0,stream>>>(F(I_CONV0+10*l+1), wCb[l], 128, DKT, 128, DKT);
  }
  for (int l = 0; l < 3; ++l) {
    k_conv_w<<<gs(256*DKT),256,0,stream>>>(F(I_DECONV0+11*l+0),  wDa[l], 256, DKT, 256, DKT);
    k_conv_w<<<gs(128*DKT),256,0,stream>>>(F(I_DECONV0+11*l+1),  wDb[l], 128, DKT, 128, DKT);
    k_conv_w<<<gs(256*DKT),256,0,stream>>>(F(I_DECONV0+11*l+10), wDc[l], 256, DKT, 256, DKT);
  }

  // ---- total_input_fea output ---------------------------------------------
  k_total_input<<<gs((long long)DNM*DOUTW),256,0,stream>>>(F(I_ATOM), F(I_NBR), idx, o_tif);

  // ---- embeddings ----------------------------------------------------------
  k_conv_x<<<gs((long long)DN1*96),256,0,stream>>>(F(I_ATOM), xaBf, DN1, 92, 96);
  gemm(xaBf, wEmbAtom, a, F(I_EMB_ATOM_B), DN1, 128, 128, 96, stream);
  k_conv_x<<<gs((long long)DNM*64),256,0,stream>>>(F(I_NBR), xnbBf, DNM, 41, 64);
  gemm(xnbBf, wEmbBond, nb, F(I_EMB_BOND_B), DNM, 64, 64, 64, stream);

  // ---- 3 conv layers -------------------------------------------------------
  for (int l = 0; l < 3; ++l) {
    const int P = I_CONV0 + 10*l;
    k_build_tot_conv<<<gs((long long)DNM*DKT),256,0,stream>>>(a, nb, idx, tot);
    // a-branch
    gemm(tot, wCa[l], big256, nullptr, DNM, 256, 256, DKT, stream);
    bn_make(big256, DNM, 256, F(P+2), F(P+3), sc1, sh1);                 // bn1a
    k_gate_asum<<<gs((long long)DN1*128),256,0,stream>>>(big256, sc1, sh1, asum);
    bn_make(asum, DN1, 128, F(P+6), F(P+7), sc2, sh2);                   // bn2a
    k_residual_sp<<<gs((long long)DN1*128),256,0,stream>>>(a, asum, sc2, sh2, (long long)DN1*128, 128);
    // b-branch
    gemm(tot, wCb[l], big256, nullptr, DNM, 128, 128, DKT, stream);
    bn_make(big256, DNM, 128, F(P+4), F(P+5), sc1, sh1);                 // bn1b
    k_gate<<<gs((long long)DNM*64),256,0,stream>>>(big256, sc1, sh1, h128, DNM, 64);
    bn_make(h128, DNM, 64, F(P+8), F(P+9), sc2, sh2);                    // bn2b
    k_residual_sp<<<gs((long long)DNM*64),256,0,stream>>>(nb, h128, sc2, sh2, (long long)DNM*64, 64);
  }

  // ---- pooling + VAE core --------------------------------------------------
  k_pool_a<<<gs((long long)DN2*DA),256,0,stream>>>(a, pooled);
  k_pool_nb<<<gs((long long)DN2*DM1*DB),256,0,stream>>>(nb, pooled);
  k_nbv<<<gs((long long)DN2*DB),256,0,stream>>>(pooled, F(I_PEMB_W), F(I_PEMB_B), nbv);
  bn_make(nbv, DN2, DB, F(I_BN_G), F(I_BN_B), sc1, sh1);
  k_build_crys<<<gs((long long)DN2*192),256,0,stream>>>(pooled, nbv, sc1, sh1, crys);
  k_vae<<<gs((long long)DN2*2),256,0,stream>>>(crys, F(I_MU_W), F(I_MU_B), F(I_LV_W), F(I_LV_B),
                                               F(I_EPS), o_mu, o_lv, o_z, zbuf);
  k_zd<<<gs((long long)DN2*192),256,0,stream>>>(zbuf, F(I_DEC_W), F(I_DEC_B), zdcat, zdn);
  k_zdn2<<<gs((long long)DN2*768),256,0,stream>>>(zdn, F(I_PEMB2_W), F(I_PEMB2_B), zdn2);
  bn_make(zdn2, (long long)DN2*DM1, DB, F(I_BN_G), F(I_BN_B), sc1, sh1); // (3600 x 64) view
  k_apply_zdn2<<<gs((long long)DN2*768),256,0,stream>>>(zdn2, sc1, sh1, zdcat);
  k_dec<<<gs((long long)DN1*896),256,0,stream>>>(a, nb, zdcat, pooled, zda2, dnb);

  // ---- zan init: broadcast(zda2) @ adjW^T ----------------------------------
  k_conv_x<<<gs((long long)DN1*128),256,0,stream>>>(zda2, zda2Bf, DN1, 128, 128);
  gemm(zda2Bf, wAdj, big256, F(I_ADJ_B), DN1, 128, 128, 128, stream);
  k_bcast_zan<<<gs((long long)DNM*128),256,0,stream>>>(big256, zan);

  // ---- 3 deconv layers -----------------------------------------------------
  for (int l = 0; l < 3; ++l) {
    const int P = I_DECONV0 + 11*l;
    k_build_tot_deconv<<<gs((long long)DNM*DKT),256,0,stream>>>(zda2, zan, dnb, tot);
    // a-branch
    gemm(tot, wDa[l], big256, nullptr, DNM, 256, 256, DKT, stream);
    bn_make(big256, DNM, 256, F(P+2), F(P+3), sc1, sh1);                 // bn1a
    k_gate_asum<<<gs((long long)DN1*128),256,0,stream>>>(big256, sc1, sh1, asum);
    bn_make(asum, DN1, 128, F(P+6), F(P+7), sc2, sh2);                   // bn2a
    k_residual_sp<<<gs((long long)DN1*128),256,0,stream>>>(zda2, asum, sc2, sh2, (long long)DN1*128, 128);
    // b-branch
    gemm(tot, wDb[l], big256, nullptr, DNM, 128, 128, DKT, stream);
    bn_make(big256, DNM, 128, F(P+4), F(P+5), sc1, sh1);                 // bn1b
    k_gate<<<gs((long long)DNM*64),256,0,stream>>>(big256, sc1, sh1, h128, DNM, 64);
    bn_make(h128, DNM, 64, F(P+8), F(P+9), sc2, sh2);                    // bn2b
    k_residual_sp<<<gs((long long)DNM*64),256,0,stream>>>(dnb, h128, sc2, sh2, (long long)DNM*64, 64);
    // n-branch (Wc; reuses bn1a / bn2a params per reference)
    gemm(tot, wDc[l], big256, nullptr, DNM, 256, 256, DKT, stream);
    bn_make(big256, DNM, 256, F(P+2), F(P+3), sc1, sh1);                 // bn1a
    k_gate<<<gs((long long)DNM*128),256,0,stream>>>(big256, sc1, sh1, h128, DNM, 128);
    bn_make(h128, DNM, 128, F(P+6), F(P+7), sc2, sh2);                   // bn2a
    k_residual_sp<<<gs((long long)DNM*128),256,0,stream>>>(zan, h128, sc2, sh2, (long long)DNM*128, 128);
  }

  // ---- final projections + emit -------------------------------------------
  float* zrLin  = h128;                              // 12000 x 92
  float* zanLin = big256;                            // 144000 x 92
  float* dnbLin = big256 + (size_t)DNM*DL;           // 144000 x 41

  k_conv_x<<<gs((long long)DN1*128),256,0,stream>>>(zda2, zda2Bf, DN1, 128, 128);
  gemm(zda2Bf, wEmbAtom2, zrLin, F(I_EMB_ATOM2_B), DN1, 92, 128, 128, stream);
  k_conv_x<<<gs((long long)DNM*128),256,0,stream>>>(zan, zanBf, DNM, 128, 128);
  gemm(zanBf, wEmbNbr, zanLin, F(I_EMB_NBR_B), DNM, 92, 128, 128, stream);
  k_conv_x<<<gs((long long)DNM*64),256,0,stream>>>(dnb, dnbBf, DNM, 64, 64);
  gemm(dnbBf, wEmbBond2, dnbLin, F(I_EMB_BOND2_B), DNM, 41, 64, 64, stream);

  k_emit<<<gs((long long)DNM*DOUTW),256,0,stream>>>(zrLin, zanLin, dnbLin, o_out);
}
